// MaskBLIP_3959959847182
// MI455X (gfx1250) — 8-bit matrix operand order repaired and checked by reading, compiled, not run on hardware
//
#include <hip/hip_runtime.h>

typedef int v8i __attribute__((ext_vector_type(8)));
typedef unsigned int u32x4 __attribute__((ext_vector_type(4)));
typedef int i32x4 __attribute__((ext_vector_type(4)));
typedef int i32x8 __attribute__((ext_vector_type(8)));

#define BATCH   8
#define HDIM    768
#define WDIM    768
#define ITERS   8
#define OUT     64
#define HALO    8
#define TILE    80
#define TSTRIDE 84
#define NWAVES  8
#define NTHREADS 256

#if __has_builtin(__builtin_amdgcn_tensor_load_to_lds) && __has_builtin(__builtin_amdgcn_s_wait_tensorcnt)
#define USE_TDM 1
#endif

__global__ __launch_bounds__(NTHREADS)
void MaskBLIP_majority_kernel(const float* __restrict__ in, float* __restrict__ out) {
    __shared__ unsigned char tile[2][TILE * TSTRIDE];
    __shared__ int sc[NWAVES][16 * 8];
#ifdef USE_TDM
    __shared__ float stage[TILE * TILE];
#endif

    const int tid    = threadIdx.x;
    const int lane   = tid & 31;
    const int waveId = tid >> 5;

    const int ox0 = blockIdx.x * OUT;
    const int oy0 = blockIdx.y * OUT;
    const int bimg = blockIdx.z;
    const int gx0 = ox0 - HALO, gy0 = oy0 - HALO;

    const int loxI = max(0, -gx0),            loyI = max(0, -gy0);
    const int hixI = min(TILE - 1, WDIM - 1 - gx0);
    const int hiyI = min(TILE - 1, HDIM - 1 - gy0);

    const float* img = in + (size_t)bimg * HDIM * WDIM;

    __builtin_prefetch(&img[min(max(gy0, 0), HDIM - 1) * WDIM +
                            min(max(gx0, 0), WDIM - 1)], 0, 1);

#ifdef USE_TDM
    const bool interior = (gx0 >= 0) & (gy0 >= 0) &
                          (gx0 + TILE <= WDIM) & (gy0 + TILE <= HDIM);
    if (interior) {
        if (tid == 0) {
            unsigned long long ga =
                (unsigned long long)(const void*)(img + (size_t)gy0 * WDIM + gx0);
            unsigned int lds_off = (unsigned int)(unsigned long long)(const void*)stage;
            u32x4 g0;
            g0[0] = 1u;
            g0[1] = lds_off;
            g0[2] = (unsigned int)(ga & 0xFFFFFFFFu);
            g0[3] = (unsigned int)((ga >> 32) & 0x01FFFFFFu) | 0x80000000u;
            const unsigned int td0 = (unsigned int)(WDIM - gx0);
            const unsigned int td1 = (unsigned int)(HDIM - gy0);
            i32x8 g1;
            g1[0] = (int)(2u << 16);
            g1[1] = (int)((td0 & 0xFFFFu) << 16);
            g1[2] = (int)(((td0 >> 16) & 0xFFFFu) | ((td1 & 0xFFFFu) << 16));
            g1[3] = (int)(((td1 >> 16) & 0xFFFFu) | ((unsigned)TILE << 16));
            g1[4] = (int)TILE;
            g1[5] = (int)WDIM;
            g1[6] = 0;
            g1[7] = 0;
            i32x4 gz  = {0, 0, 0, 0};
            i32x8 gz8 = {0, 0, 0, 0, 0, 0, 0, 0};
            __builtin_amdgcn_tensor_load_to_lds(g0, g1, gz, gz, gz8, 0);
            __builtin_amdgcn_s_wait_tensorcnt(0);
        }
        __syncthreads();
        for (int idx = tid; idx < TILE * TILE; idx += NTHREADS) {
            int ly = idx / TILE, lx = idx - ly * TILE;
            tile[0][ly * TSTRIDE + lx] = (unsigned char)(int)stage[idx];
        }
    } else
#endif
    {
        for (int idx = tid; idx < TILE * TILE; idx += NTHREADS) {
            int ly = idx / TILE, lx = idx - ly * TILE;
            int gy = min(max(gy0 + ly, 0), HDIM - 1);
            int gx = min(max(gx0 + lx, 0), WDIM - 1);
            tile[0][ly * TSTRIDE + lx] = (unsigned char)(int)img[gy * WDIM + gx];
        }
    }
    __syncthreads();

    const int mrow = lane & 15;
    const int hi   = lane >> 4;
    const int rr = mrow >> 2, cpix = mrow & 3;
    v8i Areg;
#pragma unroll
    for (int j = 0; j < 8; ++j) {
        unsigned int w = 0;
#pragma unroll
        for (int t = 0; t < 4; ++t) {
            int k = ((j >> 1) << 4) + hi * 8 + ((j & 1) << 2) + t;
            int u = (k * 43) >> 8;
            int v = k - u * 6;
            int inw = (k < 36) & (u >= rr) & (u <= rr + 2) & (v >= cpix) & (v <= cpix + 2);
            w |= (unsigned)inw << (8 * t);
        }
        Areg[j] = (int)w;
    }

    int cur = 0;
    for (int it = 0; it < ITERS; ++it) {
        const int rem = ITERS - 1 - it;
        const int lox = max(loxI, HALO - rem),            loy = max(loyI, HALO - rem);
        const int hix = min(hixI, HALO + OUT - 1 + rem);
        const int hiy = min(hiyI, HALO + OUT - 1 + rem);
        const int nbx = (hix - lox + 4) >> 2;
        const int nby = (hiy - loy + 4) >> 2;
        const int nb  = nbx * nby;
        const unsigned char* src = tile[cur];
        unsigned char*       dst = tile[cur ^ 1];

        for (int bb = waveId; bb < nb; bb += NWAVES) {
            const int bx = lox + (bb % nbx) * 4;
            const int by = loy + (bb / nbx) * 4;

            v8i Breg;
#pragma unroll
            for (int j = 0; j < 8; ++j) {
                unsigned int w = 0;
#pragma unroll
                for (int t = 0; t < 4; ++t) {
                    int k = ((j >> 1) << 4) + hi * 8 + ((j & 1) << 2) + t;
                    int u = (k * 43) >> 8;
                    int v = k - u * 6;
                    int cy = min(max(by - 1 + u, loyI), hiyI);
                    int cx = min(max(bx - 1 + v, loxI), hixI);
                    unsigned lab = src[cy * TSTRIDE + cx];
                    int hit = (k < 36) & (mrow < 8) & (lab == (unsigned)mrow);
                    w |= (unsigned)hit << (8 * t);
                }
                Breg[j] = (int)w;
            }

            v8i Cz = {0, 0, 0, 0, 0, 0, 0, 0};
            v8i D = __builtin_amdgcn_wmma_i32_16x16x64_iu8(
                        false, Areg, false, Breg, Cz, false, false);

            if (mrow < 8) {
#pragma unroll
                for (int r = 0; r < 8; ++r)
                    sc[waveId][(r + 8 * hi) * 8 + mrow] = D[r];
            }
            __builtin_amdgcn_wave_barrier();

            if (lane < 16) {
                const int base = lane * 8;
                int best = sc[waveId][base];
                int bestn = 0;
#pragma unroll
                for (int n = 1; n < 8; ++n) {
                    int cnt = sc[waveId][base + n];
                    if (cnt > best) { best = cnt; bestn = n; }
                }
                int py = by + (lane >> 2), px = bx + (lane & 3);
                if (px <= hix && py <= hiy)
                    dst[py * TSTRIDE + px] = (unsigned char)bestn;
            }
            __builtin_amdgcn_wave_barrier();
        }
        __syncthreads();
        cur ^= 1;
    }

    float* outp = out + (size_t)bimg * HDIM * WDIM;
    for (int idx = tid; idx < OUT * OUT; idx += NTHREADS) {
        int oy = idx >> 6, ox = idx & 63;
        outp[(size_t)(oy0 + oy) * WDIM + (ox0 + ox)] =
            (float)tile[0][(HALO + oy) * TSTRIDE + (HALO + ox)];
    }
}

extern "C" void kernel_launch(void* const* d_in, const int* in_sizes, int n_in,
                              void* d_out, int out_size, void* d_ws, size_t ws_size,
                              hipStream_t stream) {
    const float* clusters = (const float*)d_in[0];
    float* outp = (float*)d_out;
    (void)in_sizes; (void)n_in; (void)d_ws; (void)ws_size; (void)out_size;
    dim3 grid(WDIM / OUT, HDIM / OUT, BATCH);
    MaskBLIP_majority_kernel<<<grid, dim3(NTHREADS), 0, stream>>>(clusters, outp);
}
